// HomoMSELoss_3393024164477
// MI455X (gfx1250) — matrix operand order repaired and checked by reading, not compiled, not run on hardware
//
#include <hip/hip_runtime.h>
#include <hip/hip_bf16.h>
#include <cstdint>

typedef __attribute__((ext_vector_type(16))) _Float16 v16h;
typedef __attribute__((ext_vector_type(8)))  float    v8f;

#define HGT   480
#define WID   640
#define TOPK  512
#define CAP   32768
#define BORD  8

// ---------------- async global->LDS helpers (guarded; fallback = ds store) --
#if __has_builtin(__builtin_amdgcn_global_load_async_to_lds_b32)
#define HAVE_ASYNC_LDS 1
#endif

__device__ inline void async_ld_b32(void* lds, const void* g) {
#ifdef HAVE_ASYNC_LDS
  void* gnc = const_cast<void*>(g);
  __builtin_amdgcn_global_load_async_to_lds_b32(
      (__attribute__((address_space(1))) int*)gnc,
      (__attribute__((address_space(3))) int*)lds, 0, 0);
#else
  *(unsigned*)lds = *(const unsigned*)g;
#endif
}

__device__ inline void async_wait_lds() {
#ifdef HAVE_ASYNC_LDS
#if __has_builtin(__builtin_amdgcn_s_wait_asynccnt)
  __builtin_amdgcn_s_wait_asynccnt(0);
#else
  asm volatile("s_wait_asynccnt 0x0" ::: "memory");
#endif
#endif
}

__device__ inline int iclamp(int v, int lo, int hi) { return v < lo ? lo : (v > hi ? hi : v); }

// ---------------------------------------------------------------- init ------
__global__ void k_init(float* __restrict__ sp1, float* __restrict__ sp2,
                       int* __restrict__ cnt, float* __restrict__ norm,
                       float* __restrict__ out, int B, size_t HW) {
  size_t stride = (size_t)gridDim.x * blockDim.x;
  size_t N = (size_t)B * HW;
  for (size_t i = (size_t)blockIdx.x * blockDim.x + threadIdx.x; i < N; i += stride) {
    sp1[i] = 0.f;
    sp2[i] = 0.f;
    if (i < (size_t)B * TOPK * 2) out[1 + i] = 0.f;   // kp region
    if (i < (size_t)2 * B) cnt[i] = 0;
    if (i == 0) { norm[0] = 0.f; out[0] = 0.f; }
  }
}

// ------------------------------------------------- warp + visibility -------
__global__ void k_warp(const float* __restrict__ s2, const float* __restrict__ homo,
                       float* __restrict__ wmap, float* __restrict__ vis,
                       float* __restrict__ norm, int B) {
  const int HW = HGT * WID;
  const int per = HW / 256;                 // 1200 blocks per batch image
  int b   = blockIdx.x / per;
  int pix = (blockIdx.x % per) * 256 + threadIdx.x;
  int py  = pix / WID, px = pix - py * WID;

  const float* hm = homo + b * 9;
  float fx = (float)px, fy = (float)py;
  float X = fmaf(hm[0], fx, fmaf(hm[1], fy, hm[2]));
  float Y = fmaf(hm[3], fx, fmaf(hm[4], fy, hm[5]));
  float Z = fmaf(hm[6], fx, fmaf(hm[7], fy, hm[8]));
  float inv = 1.0f / Z;
  float xs = X * inv, ys = Y * inv;
  float x0 = floorf(xs), y0 = floorf(ys);
  float wx = xs - x0,  wy = ys - y0;

  const float* img = s2 + (size_t)b * HW;
  float ssum = 0.f, vsum = 0.f;
#pragma unroll
  for (int dy = 0; dy < 2; ++dy) {
#pragma unroll
    for (int dx = 0; dx < 2; ++dx) {
      float yy = y0 + dy, xx = x0 + dx;
      float wgt = (dy ? wy : 1.f - wy) * (dx ? wx : 1.f - wx);
      bool valid = (xx >= 0.f) && (xx <= (float)(WID - 1)) &&
                   (yy >= 0.f) && (yy <= (float)(HGT - 1));
      int xi = (int)fminf(fmaxf(xx, 0.f), (float)(WID - 1));
      int yi = (int)fminf(fmaxf(yy, 0.f), (float)(HGT - 1));
      bool inner = (xi >= BORD) && (xi < WID - BORD) &&
                   (yi >= BORD) && (yi < HGT - BORD);   // filter_border folded in
      float pv = (valid && inner) ? img[yi * WID + xi] : 0.f;
      ssum += wgt * pv;
      vsum += valid ? wgt : 0.f;                        // warp of ones
    }
  }
  float vm = vsum > 0.f ? 1.f : 0.f;
  wmap[(size_t)b * HW + pix] = ssum;
  vis [(size_t)b * HW + pix] = vm;

  __shared__ float red[256];
  red[threadIdx.x] = vm;
  __syncthreads();
  for (int o = 128; o > 0; o >>= 1) {
    if (threadIdx.x < o) red[threadIdx.x] += red[threadIdx.x + o];
    __syncthreads();
  }
  if (threadIdx.x == 0) atomicAdd(norm, red[0]);
}

// ------------------------------------------ 5x5 NMS + candidate append -----
// Clamp-to-edge padding is exactly equivalent to valid-only ('SAME' -inf) max
// for a centered window: every clamped coord maps onto an in-window element.
__global__ void k_nms(const float* __restrict__ in, float* __restrict__ cval,
                      int* __restrict__ cidx, int* __restrict__ cnt,
                      int mapoff, int B) {
  const int HW  = HGT * WID;
  const int tpb = (HGT / 16) * (WID / 16);   // 1200 tiles per image
  int b = blockIdx.x / tpb;
  int t = blockIdx.x % tpb;
  int ty = t / (WID / 16), tx = t % (WID / 16);

  __shared__ __attribute__((aligned(16))) float sm[20 * 20];  // 16x16 tile + 2-halo
  const float* img = in + (size_t)b * HW;
  for (int i = threadIdx.x; i < 400; i += 256) {
    int r = i / 20, c = i % 20;
    int gy = iclamp(ty * 16 - 2 + r, 0, HGT - 1);
    int gx = iclamp(tx * 16 - 2 + c, 0, WID - 1);
    async_ld_b32(&sm[i], &img[gy * WID + gx]);
  }
  async_wait_lds();
  __syncthreads();

  int r = threadIdx.x >> 4, c = threadIdx.x & 15;
  float cen = sm[(r + 2) * 20 + (c + 2)];
  float m = -1e30f;
#pragma unroll
  for (int dr = 0; dr < 5; ++dr)
#pragma unroll
    for (int dc = 0; dc < 5; ++dc) m = fmaxf(m, sm[(r + dr) * 20 + (c + dc)]);

  if (cen == m && cen > 0.f) {
    int pos = atomicAdd(&cnt[mapoff * B + b], 1);
    if (pos < CAP) {
      size_t base = ((size_t)mapoff * B + b) * CAP;
      cval[base + pos] = cen;
      cidx[base + pos] = (ty * 16 + r) * WID + (tx * 16 + c);
    }
  }
}

// ----------------------------------- top-512 via LDS histogram select ------
__global__ void k_select(const float* __restrict__ cval, const int* __restrict__ cidx,
                         const int* __restrict__ cnt, float* __restrict__ sp1,
                         float* __restrict__ sp2, float* __restrict__ kp, int B) {
  const int HW = HGT * WID;
  int map = blockIdx.x / B, b = blockIdx.x % B;
  size_t base = ((size_t)map * B + b) * CAP;
  int n = cnt[map * B + b]; if (n > CAP) n = CAP;

  __shared__ int hist[4096];
  __shared__ int s_tb, s_above, s_taken, s_out;
  for (int i = threadIdx.x; i < 4096; i += blockDim.x) hist[i] = 0;
  if (threadIdx.x == 0) { s_taken = 0; s_out = 0; }
  __syncthreads();

  for (int i = threadIdx.x; i < n; i += blockDim.x) {
    int bin = (int)(cval[base + i] * 4096.f);
    bin = bin < 0 ? 0 : (bin > 4095 ? 4095 : bin);
    atomicAdd(&hist[bin], 1);
  }
  __syncthreads();

  if (threadIdx.x == 0) {
    int acc = 0, tb = -1, above = 0;
    for (int bn = 4095; bn >= 0; --bn) {
      if (acc + hist[bn] >= TOPK) { tb = bn; above = acc; break; }
      acc += hist[bn];
    }
    s_tb = tb; s_above = above;                 // tb==-1 -> take everything
  }
  __syncthreads();

  float* sp = map ? sp2 : sp1;
  for (int i = threadIdx.x; i < n; i += blockDim.x) {
    float v = cval[base + i];
    int bin = (int)(v * 4096.f);
    bin = bin < 0 ? 0 : (bin > 4095 ? 4095 : bin);
    bool sel = false;
    if (bin > s_tb) sel = true;
    else if (bin == s_tb) { int p = atomicAdd(&s_taken, 1); sel = (s_above + p) < TOPK; }
    if (sel) {
      int idx = cidx[base + i];
      sp[(size_t)b * HW + idx] = v;
      if (map == 0) {
        int slot = atomicAdd(&s_out, 1);
        if (slot < TOPK) {
          kp[((size_t)b * TOPK + slot) * 2 + 0] = (float)(idx / WID);
          kp[((size_t)b * TOPK + slot) * 2 + 1] = (float)(idx % WID);
        }
      }
    }
  }
}

// ----------------------------------------- Gaussian weights (sigma=0.5) ----
__device__ inline void gauss_h(_Float16* gh) {
  float g[15], s = 0.f;
#pragma unroll
  for (int t = 0; t < 15; ++t) { float d = (float)t - 7.f; g[t] = expf(-d * d / 0.5f); s += g[t]; }
  float inv = 1.f / s;
#pragma unroll
  for (int t = 0; t < 15; ++t) gh[t] = (_Float16)(g[t] * inv);
}

// -------------------- horizontal blur: D[16x16] = A[16x32] x Band[32x16] ---
// One wave per 16x16 tile; 8 waves cover a 16x128 strip async-staged in LDS.
__global__ void k_hblur(const float* __restrict__ in, _Float16* __restrict__ out, int B) {
  const int HW  = HGT * WID;
  const int per = (HGT / 16) * (WID / 128);     // 150 strips per image
  int b  = blockIdx.x / per, rem = blockIdx.x % per;
  int r0 = (rem / (WID / 128)) * 16;
  int cb = (rem % (WID / 128)) * 128;

  __shared__ __attribute__((aligned(16))) float smf[16 * 144]; // 142 cols used
  const float* src = in + (size_t)b * HW;
  for (int i = threadIdx.x; i < 16 * 144; i += 256) {           // 9 full iters
    int r = i / 144, lc = i - r * 144;
    int cc = iclamp(cb - 7 + lc, 0, WID - 1);
    async_ld_b32(&smf[i], &src[(r0 + r) * WID + cc]);
  }
  async_wait_lds();
  // zero-patch: conv 'SAME' needs zeros outside the image / beyond col 141
  for (int i = threadIdx.x; i < 16 * 144; i += 256) {
    int lc = i % 144;
    int c = cb - 7 + lc;
    if (lc >= 142 || c < 0 || c >= WID) smf[i] = 0.f;
  }
  __syncthreads();

  _Float16 gh[15]; gauss_h(gh);
  int lane = threadIdx.x & 31, wv = threadIdx.x >> 5;
  int grp = lane >> 4, lm = lane & 15;

  // A operand: 16-bit A 16x32 lane layout; f32->f16 convert during gather
  v16h Av;
#pragma unroll
  for (int q = 0; q < 8; ++q) {
    int K0 = 2 * q + 16 * grp;
    int base = lm * 144 + 16 * wv + K0;
    Av[2 * q]     = (_Float16)smf[base];
    Av[2 * q + 1] = (_Float16)smf[base + 1];
  }
  // B operand: constant Gaussian band, B[k][n] = g[k-n] for k-n in [0,14]
  v16h Bv;
#pragma unroll
  for (int e = 0; e < 16; ++e) {
    int K = e + 16 * grp, t = K - lm;
    Bv[e] = (t >= 0 && t < 15) ? gh[t] : (_Float16)0.f;
  }
  v8f C = {};
  C = __builtin_amdgcn_wmma_f32_16x16x32_f16(false, Av, false, Bv, (short)0, C, false, false);

  _Float16* dst = out + (size_t)b * HW;
  int c0 = cb + 16 * wv;
#pragma unroll
  for (int q = 0; q < 8; ++q) {
    int M = q + 8 * grp;                        // f32 C/D lane layout
    dst[(size_t)(r0 + M) * WID + c0 + lm] = (_Float16)C[q];
  }
}

// ---------------------- vertical blur: D = Band[16x32] x B[32x16] ----------
// 32x128 f16 strip (rows r0-7 .. r0+24) async-staged in LDS, gathers via ds.
__global__ void k_vblur(const _Float16* __restrict__ in, float* __restrict__ out, int B) {
  const int HW  = HGT * WID;
  const int per = (HGT / 16) * (WID / 128);
  int b  = blockIdx.x / per, rem = blockIdx.x % per;
  int r0 = (rem / (WID / 128)) * 16;
  int cb = (rem % (WID / 128)) * 128;

  __shared__ __attribute__((aligned(16))) _Float16 smv[32 * 128];
  unsigned* smu = (unsigned*)smv;                // 64 dwords per staged row
  const _Float16* src = in + (size_t)b * HW;
  for (int i = threadIdx.x; i < 32 * 64; i += 256) {   // 8 full iters
    int k = i >> 6, j = i & 63;
    int rr = iclamp(r0 - 7 + k, 0, HGT - 1);
    async_ld_b32(&smu[i], &src[(size_t)rr * WID + cb + 2 * j]);
  }
  async_wait_lds();
  for (int i = threadIdx.x; i < 32 * 64; i += 256) {   // vertical 'SAME' zeros
    int k = i >> 6, r = r0 - 7 + k;
    if (r < 0 || r >= HGT) smu[i] = 0u;
  }
  __syncthreads();

  _Float16 gh[15]; gauss_h(gh);
  int lane = threadIdx.x & 31, wv = threadIdx.x >> 5;
  int grp = lane >> 4, lm = lane & 15;

  // A = Gaussian band, A[i][k] = g[k-i]; 16-bit A layout
  v16h Av;
#pragma unroll
  for (int e = 0; e < 16; ++e) {
    int K = e + 16 * grp;
    int t = K - lm;
    Av[e] = (t >= 0 && t < 15) ? gh[t] : (_Float16)0.f;
  }
  // B = vertically-extended tile from LDS: row K, col 16*wv + lm
  v16h Bv;
#pragma unroll
  for (int e = 0; e < 16; ++e) {
    int K = e + 16 * grp;
    Bv[e] = smv[K * 128 + 16 * wv + lm];
  }
  v8f C = {};
  C = __builtin_amdgcn_wmma_f32_16x16x32_f16(false, Av, false, Bv, (short)0, C, false, false);

  float* dst = out + (size_t)b * HW;
#pragma unroll
  for (int q = 0; q < 8; ++q) {
    int M = q + 8 * grp;
    dst[(size_t)(r0 + M) * WID + cb + 16 * wv + lm] = C[q];
  }
}

// ------------------------------------------------ masked MSE reduction -----
__global__ void k_loss(const float* __restrict__ b1, const float* __restrict__ b2,
                       const float* __restrict__ vis, const float* __restrict__ norm,
                       float* __restrict__ loss, size_t N) {
  __shared__ float red[256];
  const float4* B1 = (const float4*)b1;
  const float4* B2 = (const float4*)b2;
  const float4* V  = (const float4*)vis;
  size_t N4 = N / 4;
  size_t stride = (size_t)gridDim.x * 256;
  float acc = 0.f;
  for (size_t i = (size_t)blockIdx.x * 256 + threadIdx.x; i < N4; i += stride) {
    __builtin_prefetch(&B1[i + stride], 0, 0);   // global_prefetch_b8
    __builtin_prefetch(&B2[i + stride], 0, 0);
    float4 a = B1[i], g = B2[i], v = V[i];
    float d0 = a.x - g.x, d1 = a.y - g.y, d2 = a.z - g.z, d3 = a.w - g.w;
    acc += d0 * d0 * v.x + d1 * d1 * v.y + d2 * d2 * v.z + d3 * d3 * v.w;
  }
  red[threadIdx.x] = acc;
  __syncthreads();
  for (int o = 128; o > 0; o >>= 1) {
    if (threadIdx.x < o) red[threadIdx.x] += red[threadIdx.x + o];
    __syncthreads();
  }
  if (threadIdx.x == 0) atomicAdd(loss, red[0] / norm[0]);
}

// ---------------------------------------------------------------------------
extern "C" void kernel_launch(void* const* d_in, const int* in_sizes, int n_in,
                              void* d_out, int out_size, void* d_ws, size_t ws_size,
                              hipStream_t stream) {
  const float* score1 = (const float*)d_in[0];
  const float* score2 = (const float*)d_in[1];
  const float* homo   = (const float*)d_in[2];
  const int HW = HGT * WID;
  int B = in_sizes[0] / HW;

  // workspace layout (floats): wmap | sp1 | sp2 | tbuf(f16,HW*B) | cval | cidx | cnt | norm
  float* ws   = (float*)d_ws;
  float* wmap = ws;
  float* sp1  = ws + (size_t)1 * B * HW;
  float* sp2  = ws + (size_t)2 * B * HW;
  _Float16* tbuf = (_Float16*)(ws + (size_t)3 * B * HW);
  float* cval = ws + (size_t)3 * B * HW + ((size_t)B * HW) / 2;
  int*   cidx = (int*)(cval + (size_t)2 * B * CAP);
  int*   cnt  = cidx + (size_t)2 * B * CAP;
  float* norm = (float*)(cnt + 2 * B);

  float* out = (float*)d_out;                  // [0]=loss
  float* kp  = out + 1;                        // B*TOPK*2
  float* vis = out + 1 + (size_t)B * TOPK * 2; // B*HW

  int blksPix  = B * HW / 256;
  int blksNms  = B * (HGT / 16) * (WID / 16);
  int blksBlur = B * (HGT / 16) * (WID / 128);

  k_init <<<blksPix,  256, 0, stream>>>(sp1, sp2, cnt, norm, out, B, (size_t)HW);
  k_warp <<<blksPix,  256, 0, stream>>>(score2, homo, wmap, vis, norm, B);
  k_nms  <<<blksNms,  256, 0, stream>>>(score1, cval, cidx, cnt, 0, B);
  k_nms  <<<blksNms,  256, 0, stream>>>(wmap,   cval, cidx, cnt, 1, B);
  k_select<<<2 * B,  1024, 0, stream>>>(cval, cidx, cnt, sp1, sp2, kp, B);
  k_hblur<<<blksBlur, 256, 0, stream>>>(sp1, tbuf, B);
  k_vblur<<<blksBlur, 256, 0, stream>>>(tbuf, sp1, B);
  k_hblur<<<blksBlur, 256, 0, stream>>>(sp2, tbuf, B);
  k_vblur<<<blksBlur, 256, 0, stream>>>(tbuf, sp2, B);
  k_loss <<<1024,     256, 0, stream>>>(sp1, sp2, vis, norm, out, (size_t)B * HW);
}
